// GIN_45921790329151
// MI455X (gfx1250) — matrix operand order repaired and checked by reading, not compiled, not run on hardware
//
#include <hip/hip_runtime.h>
#include <hip/hip_bf16.h>

// ---------------------------------------------------------------------------
// GIN forward for MI455X (gfx1250, wave32, WMMA).
//   N=100000 nodes, E=1600000 edges, FX=128, MAXZ=100, HID=128, NL=3, G=512
//   h0 = [one_hot(z), one_hot(z), x]  -> [N, 328] (padded to 352 for K%32==0)
//   per layer: agg = scatter_add(h[src] -> dst); m = relu(relu((h+agg)W1+b1)W2+b2)
//              h' = batchnorm(m) (training-mode, biased var)  -> hcat slice
//   out = relu(mean_pool(hcat) @ Wl1 + bl1) @ Wl2 + bl2
// GEMMs: v_wmma_f32_16x16x32_bf16; B staged in LDS, double-buffered ds reads.
// ---------------------------------------------------------------------------

#define GIN_N   100000
#define GIN_E   1600000
#define GIN_G   512
#define GIN_HID 128
#define GIN_K0P 352      // 328 padded to multiple of 32
#define GIN_K0R 328
#define GIN_CAT 384      // NL*HID

typedef __bf16 bf16_t;
typedef __attribute__((ext_vector_type(16))) __bf16 v16bf;
typedef __attribute__((ext_vector_type(8)))  __bf16 v8bf;
typedef __attribute__((ext_vector_type(8)))  float  v8f;

__device__ __forceinline__ unsigned short gin_f2bf(float f) {
    unsigned int u = __float_as_uint(f);
    unsigned int r = u + 0x7FFFu + ((u >> 16) & 1u);  // round-to-nearest-even
    return (unsigned short)(r >> 16);
}

// --------------------------------------------------------------------------
// Build h0 = [z_onehot | z_onehot | x | zero-pad] as [N, 352] f32
// --------------------------------------------------------------------------
__global__ void gin_build_h0(const float* __restrict__ x, const int* __restrict__ z,
                             float* __restrict__ h0) {
    long long idx = (long long)blockIdx.x * blockDim.x + threadIdx.x;
    long long total = (long long)GIN_N * GIN_K0P;
    if (idx >= total) return;
    int i = (int)(idx / GIN_K0P);
    int c = (int)(idx % GIN_K0P);
    float v = 0.0f;
    if (c < 200) {
        v = ((c % 100) == z[i]) ? 1.0f : 0.0f;
    } else if (c < GIN_K0R) {
        v = x[(size_t)i * 128 + (c - 200)];
    }
    h0[idx] = v;
}

// --------------------------------------------------------------------------
// Edge scatter-add: agg[dst] += H[src]   (float4 chunks, f32 atomics; L2-resident)
// --------------------------------------------------------------------------
__global__ void gin_edge_agg(const int* __restrict__ src, const int* __restrict__ dst,
                             const float* __restrict__ H, int ldh,
                             float* __restrict__ agg, int ldo, int K) {
    long long total = (long long)GIN_E * (K >> 2);
    long long stride = (long long)gridDim.x * blockDim.x;
    int nc = K >> 2;
    for (long long idx = (long long)blockIdx.x * blockDim.x + threadIdx.x;
         idx < total; idx += stride) {
        int e  = (int)(idx / nc);
        int c4 = (int)(idx % nc) << 2;
        int s = src[e], d = dst[e];
        const float4 v = *(const float4*)(H + (size_t)s * ldh + c4);
        float* op = agg + (size_t)d * ldo + c4;
        atomicAdd(op + 0, v.x);
        atomicAdd(op + 1, v.y);
        atomicAdd(op + 2, v.z);
        atomicAdd(op + 3, v.w);
    }
}

// --------------------------------------------------------------------------
// m = bf16(H + agg)   -> [N, K] row-major bf16 (WMMA A operand)
// --------------------------------------------------------------------------
__global__ void gin_add_conv(const float* __restrict__ H, int ldh,
                             const float* __restrict__ agg, int lda,
                             unsigned short* __restrict__ mb, int K) {
    long long idx = (long long)blockIdx.x * blockDim.x + threadIdx.x;
    long long total = (long long)GIN_N * K;
    if (idx >= total) return;
    int i = (int)(idx / K);
    int c = (int)(idx % K);
    float v = H[(size_t)i * ldh + c] + agg[(size_t)i * lda + c];
    mb[idx] = gin_f2bf(v);
}

// --------------------------------------------------------------------------
// Pack W [Kreal,128] f32 -> bf16 WMMA B fragments (zero-pad K to Kp).
// kt-major layout: frag[((kt*8 + ct)*32 + lane)*16 + i]
//   lane holds column ct*16+(lane&15); i indexes k = kt*32 + 16*(i>>3) + 8*(lane>>4) + (i&7)
// --------------------------------------------------------------------------
__global__ void gin_pack_w(const float* __restrict__ W,
                           unsigned short* __restrict__ out, int Kreal, int Kp) {
    int nkt = Kp >> 5;
    int total = 8 * nkt * 32 * 16;
    int idx = blockIdx.x * blockDim.x + threadIdx.x;
    if (idx >= total) return;
    int i    = idx & 15;
    int lane = (idx >> 4) & 31;
    int rest = idx >> 9;      // = kt*8 + ct
    int ct = rest & 7;
    int kt = rest >> 3;
    int k   = kt * 32 + ((i >> 3) * 16) + (lane >> 4) * 8 + (i & 7);
    int col = ct * 16 + (lane & 15);
    float v = (k < Kreal) ? W[(size_t)k * 128 + col] : 0.0f;
    out[idx] = gin_f2bf(v);
}

// --------------------------------------------------------------------------
// WMMA GEMM: C[M,128] = relu(A[M,Kp](bf16) @ Bpack + bias)
// One wave = 16-row strip x all 8 column tiles (8 f32 accum tiles = 64 VGPRs).
// Packed B (Kp*128 bf16, 32-88 KB) staged into LDS once per block.
// Inner loop: branchless A prefetch (global), double-buffered B fragments so
// each tile's ds_load_b128 pair is in flight underneath the previous WMMA,
// then 8 v_wmma_f32_16x16x32_bf16 per k-step.
// Rows >= M are clamped on load (their C rows are never stored; WMMA rows are
// independent, so garbage cannot pollute valid rows).
// --------------------------------------------------------------------------
template <bool WRITE_F32, bool WRITE_BF16>
__global__ void gin_wmma_gemm(const unsigned short* __restrict__ Au,
                              const unsigned short* __restrict__ Bp,
                              const float* __restrict__ bias,
                              float* __restrict__ Cf,
                              unsigned short* __restrict__ Cbf,
                              int M, int Kp) {
    extern __shared__ unsigned short ldsb[];   // Kp*128 bf16
    const int nkt = Kp >> 5;

    // ---- cooperative stage: packed B -> LDS (128-thread block, uint4 chunks)
    {
        const uint4* s = (const uint4*)Bp;
        uint4*       d = (uint4*)ldsb;
        int n16 = (8 * nkt * 32 * 16) >> 3;    // ushorts / 8 per uint4
        for (int t = threadIdx.x; t < n16; t += blockDim.x) d[t] = s[t];
    }
    __syncthreads();

    int wave = (int)((blockIdx.x * (long long)blockDim.x + threadIdx.x) >> 5);
    int lane = threadIdx.x & 31;
    int row_base = wave * 16;
    if (row_base >= M) return;           // wave-uniform exit (EXEC stays all-ones)

    v8f acc[8];
#pragma unroll
    for (int ct = 0; ct < 8; ++ct)
#pragma unroll
        for (int r = 0; r < 8; ++r) acc[ct][r] = 0.0f;

    int arow = row_base + (lane & 15);
    if (arow >= M) arow = M - 1;         // clamp: harmless, see note above
    const int kb = (lane >> 4) * 8;
    const unsigned short* Arow = Au + (size_t)arow * Kp;
    const unsigned short* lbase = ldsb + lane * 16;   // + (kt*8+ct)*512 elems

    // software-pipelined A fragment (2 x 16B contiguous loads per lane)
    v8bf alo = *(const v8bf*)(Arow + kb);
    v8bf ahi = *(const v8bf*)(Arow + 16 + kb);

    for (int kt = 0; kt < nkt; ++kt) {
        v16bf afrag;
#pragma unroll
        for (int i = 0; i < 8; ++i) { afrag[i] = alo[i]; afrag[i + 8] = ahi[i]; }

        // branchless prefetch of next k-step's A (clamped re-read on last step)
        int ktn = (kt + 1 < nkt) ? kt + 1 : kt;
        alo = *(const v8bf*)(Arow + ktn * 32 + kb);
        ahi = *(const v8bf*)(Arow + ktn * 32 + 16 + kb);

        // double-buffered B fragments: loads for ct+1 issue before WMMA(ct)
        const unsigned short* kbase = lbase + (size_t)kt * 8 * 512;
        v16bf bcur = *(const v16bf*)(kbase);
#pragma unroll
        for (int ct = 0; ct < 8; ++ct) {
            v16bf bnext;
            if (ct < 7) bnext = *(const v16bf*)(kbase + (size_t)(ct + 1) * 512);
            acc[ct] = __builtin_amdgcn_wmma_f32_16x16x32_bf16(
                false, afrag, false, bcur, (short)0, acc[ct], false, false);
            if (ct < 7) bcur = bnext;
        }
    }

    // C layout: VGPR r -> row = row_base + (lane>=16 ? 8 : 0) + r, col = ct*16 + (lane&15)
    int half = lane >> 4;
    int ncol = lane & 15;
#pragma unroll
    for (int ct = 0; ct < 8; ++ct) {
        int col = ct * 16 + ncol;
        float b = bias[col];
#pragma unroll
        for (int r = 0; r < 8; ++r) {
            int row = row_base + half * 8 + r;
            if (row < M) {
                float v = fmaxf(acc[ct][r] + b, 0.0f);
                if (WRITE_F32)  Cf[(size_t)row * 128 + col] = v;
                if (WRITE_BF16) Cbf[(size_t)row * 128 + col] = gin_f2bf(v);
            }
        }
    }
}

// --------------------------------------------------------------------------
// BatchNorm stats: per-channel mean + rsqrt(var+eps) over N rows
// --------------------------------------------------------------------------
__global__ void gin_bn_stats(const float* __restrict__ m2, float* __restrict__ stats) {
    int c = blockIdx.x;  // 0..127
    float s = 0.0f, s2 = 0.0f;
    for (int i = threadIdx.x; i < GIN_N; i += blockDim.x) {
        float v = m2[(size_t)i * 128 + c];
        s += v; s2 += v * v;
    }
    __shared__ float sh[256], sh2[256];
    sh[threadIdx.x] = s; sh2[threadIdx.x] = s2;
    __syncthreads();
    for (int off = blockDim.x >> 1; off > 0; off >>= 1) {
        if (threadIdx.x < off) {
            sh[threadIdx.x]  += sh[threadIdx.x + off];
            sh2[threadIdx.x] += sh2[threadIdx.x + off];
        }
        __syncthreads();
    }
    if (threadIdx.x == 0) {
        float mu  = sh[0] / (float)GIN_N;
        float var = sh2[0] / (float)GIN_N - mu * mu;
        stats[c]       = mu;
        stats[128 + c] = rsqrtf(var + 1e-5f);
    }
}

__global__ void gin_bn_apply(const float* __restrict__ m2, const float* __restrict__ stats,
                             const float* __restrict__ g, const float* __restrict__ be,
                             float* __restrict__ hcat, int loff) {
    long long idx = (long long)blockIdx.x * blockDim.x + threadIdx.x;
    long long total = (long long)GIN_N * 128;
    if (idx >= total) return;
    int i = (int)(idx / 128);
    int c = (int)(idx % 128);
    float v = (m2[idx] - stats[c]) * stats[128 + c] * g[c] + be[c];
    hcat[(size_t)i * GIN_CAT + loff + c] = v;
}

// --------------------------------------------------------------------------
// Graph pooling + head MLP
// --------------------------------------------------------------------------
__global__ void gin_count(const int* __restrict__ batch, float* __restrict__ counts) {
    int i = blockIdx.x * blockDim.x + threadIdx.x;
    if (i < GIN_N) atomicAdd(&counts[batch[i]], 1.0f);
}

__global__ void gin_pool(const int* __restrict__ batch, const float* __restrict__ hcat,
                         float* __restrict__ pooled) {
    long long idx = (long long)blockIdx.x * blockDim.x + threadIdx.x;
    long long total = (long long)GIN_N * GIN_CAT;
    if (idx >= total) return;
    int i = (int)(idx / GIN_CAT);
    int c = (int)(idx % GIN_CAT);
    atomicAdd(&pooled[(size_t)batch[i] * GIN_CAT + c], hcat[idx]);
}

__global__ void gin_head(const float* __restrict__ pooled, const float* __restrict__ counts,
                         const float* __restrict__ Wl1, const float* __restrict__ bl1,
                         const float* __restrict__ Wl2, const float* __restrict__ bl2,
                         float* __restrict__ out) {
    int g = blockIdx.x;    // 0..511
    int j = threadIdx.x;   // 0..127
    float inv = 1.0f / fmaxf(counts[g], 1.0f);
    const float* pr = pooled + (size_t)g * GIN_CAT;
    float acc = bl1[j];
    for (int c = 0; c < GIN_CAT; ++c)
        acc += (pr[c] * inv) * Wl1[(size_t)c * 128 + j];
    acc = fmaxf(acc, 0.0f);
    __shared__ float sh[128];
    sh[j] = acc * Wl2[j];
    __syncthreads();
    for (int off = 64; off > 0; off >>= 1) {
        if (j < off) sh[j] += sh[j + off];
        __syncthreads();
    }
    if (j == 0) out[g] = sh[0] + bl2[0];
}

// --------------------------------------------------------------------------
// Host orchestration
// --------------------------------------------------------------------------
static inline long long gin_blocks(long long total, int bs) { return (total + bs - 1) / bs; }

extern "C" void kernel_launch(void* const* d_in, const int* in_sizes, int n_in,
                              void* d_out, int out_size, void* d_ws, size_t ws_size,
                              hipStream_t stream) {
    const float* x          = (const float*)d_in[0];
    const int*   z          = (const int*)d_in[1];
    const int*   edge_index = (const int*)d_in[2];
    const int*   batch      = (const int*)d_in[3];
    const int*   src = edge_index;
    const int*   dst = edge_index + GIN_E;

    const float* W1[3] = { (const float*)d_in[4],  (const float*)d_in[10], (const float*)d_in[16] };
    const float* b1[3] = { (const float*)d_in[5],  (const float*)d_in[11], (const float*)d_in[17] };
    const float* W2[3] = { (const float*)d_in[6],  (const float*)d_in[12], (const float*)d_in[18] };
    const float* b2[3] = { (const float*)d_in[7],  (const float*)d_in[13], (const float*)d_in[19] };
    const float* gg[3] = { (const float*)d_in[8],  (const float*)d_in[14], (const float*)d_in[20] };
    const float* be[3] = { (const float*)d_in[9],  (const float*)d_in[15], (const float*)d_in[21] };
    const float* Wl1 = (const float*)d_in[22];
    const float* bl1 = (const float*)d_in[23];
    const float* Wl2 = (const float*)d_in[24];
    const float* bl2 = (const float*)d_in[25];
    float* out = (float*)d_out;

    // ---- workspace carve-out (256B aligned slices) ----
    char* ws = (char*)d_ws;
    size_t off = 0;
    auto carve = [&](size_t bytes) -> void* {
        void* p = ws + off;
        off += (bytes + 255) & ~(size_t)255;
        return p;
    };
    float*          h0    = (float*)carve((size_t)GIN_N * GIN_K0P * 4);   // 140.8 MB
    float*          agg   = (float*)carve((size_t)GIN_N * GIN_K0P * 4);   // 140.8 MB
    unsigned short* mbf   = (unsigned short*)carve((size_t)GIN_N * GIN_K0P * 2); // 70.4 MB
    unsigned short* t1    = (unsigned short*)carve((size_t)GIN_N * 128 * 2);     // 25.6 MB
    float*          hcat  = (float*)carve((size_t)GIN_N * GIN_CAT * 4);   // 153.6 MB
    unsigned short* wp1   = (unsigned short*)carve((size_t)GIN_K0P * 128 * 2);
    unsigned short* wp2   = (unsigned short*)carve((size_t)128 * 128 * 2);
    float*          stats = (float*)carve(2 * 128 * 4);
    float*          pooled= (float*)carve((size_t)GIN_G * GIN_CAT * 4);
    float*          counts= (float*)carve((size_t)GIN_G * 4);
    float*          m2    = agg;   // alias: agg is dead after add_conv, m2 written after

    const int BS = 256;

    // ---- build input features ----
    gin_build_h0<<<(int)gin_blocks((long long)GIN_N * GIN_K0P, BS), BS, 0, stream>>>(x, z, h0);

    hipMemsetAsync(pooled, 0, (size_t)GIN_G * GIN_CAT * 4, stream);
    hipMemsetAsync(counts, 0, (size_t)GIN_G * 4, stream);

    const int gemm_blocks = (int)(((GIN_N + 15) / 16 + 3) / 4);  // 4 waves/block

    for (int l = 0; l < 3; ++l) {
        const int Kp    = (l == 0) ? GIN_K0P : 128;
        const int Kreal = (l == 0) ? GIN_K0R : 128;
        const float* Hin = (l == 0) ? h0 : (hcat + (l - 1) * 128);
        const int ldh    = (l == 0) ? GIN_K0P : GIN_CAT;
        const size_t lds1 = (size_t)Kp * 128 * 2;   // GEMM1 B tile bytes
        const size_t lds2 = (size_t)128 * 128 * 2;  // GEMM2 B tile bytes

        // agg = 0; agg[dst] += Hin[src]
        hipMemsetAsync(agg, 0, (size_t)GIN_N * Kp * 4, stream);
        gin_edge_agg<<<(int)gin_blocks((long long)GIN_E * (Kp >> 2), BS), BS, 0, stream>>>(
            src, dst, Hin, ldh, agg, Kp, Kp);

        // m = bf16(Hin + agg)
        gin_add_conv<<<(int)gin_blocks((long long)GIN_N * Kp, BS), BS, 0, stream>>>(
            Hin, ldh, agg, Kp, mbf, Kp);

        // t1 = relu(m @ W1 + b1)  (bf16 out, feeds GEMM2)
        gin_pack_w<<<(int)gin_blocks(8LL * (Kp >> 5) * 512, BS), BS, 0, stream>>>(
            W1[l], wp1, Kreal, Kp);
        gin_wmma_gemm<false, true><<<gemm_blocks, 128, lds1, stream>>>(
            mbf, wp1, b1[l], (float*)nullptr, t1, GIN_N, Kp);

        // m2 = relu(t1 @ W2 + b2)  (f32 out, feeds BN)
        gin_pack_w<<<(int)gin_blocks(8LL * 4 * 512, BS), BS, 0, stream>>>(
            W2[l], wp2, 128, 128);
        gin_wmma_gemm<true, false><<<gemm_blocks, 128, lds2, stream>>>(
            t1, wp2, b2[l], m2, (unsigned short*)nullptr, GIN_N, 128);

        // BatchNorm -> hcat[:, l*128 : (l+1)*128]
        gin_bn_stats<<<128, 256, 0, stream>>>(m2, stats);
        gin_bn_apply<<<(int)gin_blocks((long long)GIN_N * 128, BS), BS, 0, stream>>>(
            m2, stats, gg[l], be[l], hcat, l * 128);
    }

    // ---- pooling + head ----
    gin_count<<<(int)gin_blocks(GIN_N, BS), BS, 0, stream>>>(batch, counts);
    gin_pool<<<(int)gin_blocks((long long)GIN_N * GIN_CAT, BS), BS, 0, stream>>>(
        batch, hcat, pooled);
    gin_head<<<GIN_G, 128, 0, stream>>>(pooled, counts, Wl1, bl1, Wl2, bl2, out);
}
